// DynamicHyperNetwork_74904229642826
// MI455X (gfx1250) — matrix operand order repaired and checked by reading, not compiled, not run on hardware
//
#include <hip/hip_runtime.h>

// ---------------------------------------------------------------------------
// DynamicHyperNetwork on MI455X (gfx1250): bf16 WMMA for all GEMM stages,
// async global->LDS staging (ASYNCcnt) on the serial GRU critical path.
// B=64, S=1024, H=256, NF=C=256, K=7, POOL=3.
// ---------------------------------------------------------------------------

typedef __attribute__((ext_vector_type(16))) __bf16 v16bf;
typedef __attribute__((ext_vector_type(8)))  __bf16 v8bf;
typedef __attribute__((ext_vector_type(8)))  float  v8f;

#define BB   64
#define SS   1024
#define HH   256
#define NFF  256
#define KK   7
#define H3   768        // 3*H
#define KC   1792       // K*C (im2col depth) == K*NF (hyper_b cols)
#define POOLED 341      // floor(1024/3)

__device__ __forceinline__ v16bf cat8(v8bf lo, v8bf hi) {
  return __builtin_shufflevector(lo, hi, 0,1,2,3,4,5,6,7,8,9,10,11,12,13,14,15);
}

__device__ __forceinline__ v8bf zero8bf() {
  v8bf z;
#pragma unroll
  for (int i = 0; i < 8; ++i) z[i] = (__bf16)0.0f;
  return z;
}

__device__ __forceinline__ v8f splat8(float v) {
  v8f r;
#pragma unroll
  for (int i = 0; i < 8; ++i) r[i] = v;
  return r;
}

// 16-bit A fragment (16x32, MxK): lane holds row (lane&15); chunk select by lane>>4.
__device__ __forceinline__ v16bf load_a_frag(const __bf16* rowp, int kbase, int khalf) {
  v8bf lo = *reinterpret_cast<const v8bf*>(rowp + kbase + khalf * 8);
  v8bf hi = *reinterpret_cast<const v8bf*>(rowp + kbase + 16 + khalf * 8);
  return cat8(lo, hi);
}

// 16-bit B fragment (32x16, KxN): lane holds column; chunk select by lane>>4, same k order as A (needs [N][K] layout).
__device__ __forceinline__ v16bf load_b_frag(const __bf16* colp, int kbase, int khalf) {
  v8bf lo = *reinterpret_cast<const v8bf*>(colp + kbase + khalf * 8);
  v8bf hi = *reinterpret_cast<const v8bf*>(colp + kbase + 16 + khalf * 8);
  return cat8(lo, hi);
}

__device__ __forceinline__ v8f wmma_bf16(v16bf a, v16bf b, v8f c) {
  return __builtin_amdgcn_wmma_f32_16x16x32_bf16(false, a, false, b, (short)0, c, false, false);
}

__device__ __forceinline__ float sigmoidf_(float x) { return 1.0f / (1.0f + __expf(-x)); }

// ---------------------------------------------------------------------------
// Prep kernels.
// ---------------------------------------------------------------------------
__global__ void k_cast_bf16(const float* __restrict__ in, __bf16* __restrict__ out, long n) {
  long i = (long)blockIdx.x * blockDim.x + threadIdx.x;
  if (i < n) out[i] = (__bf16)in[i];
}

__global__ void k_transpose_bf16(const float* __restrict__ in, __bf16* __restrict__ out,
                                 int rows, int cols) {
  int c = blockIdx.x * 16 + threadIdx.x;
  int r = blockIdx.y * 16 + threadIdx.y;
  if (r < rows && c < cols) out[(size_t)c * rows + r] = (__bf16)in[(size_t)r * cols + c];
}

// ---------------------------------------------------------------------------
// Stage 1: xp[dir][b][s][0:768] = x[b][sx] @ Wk[dir] + b0[dir]  (sx reversed for bwd)
// grid (4096, 1, 2), block 512 (16 waves x 3 N-tiles = 48). A tile staged in LDS
// once per block (8 KB); A fragments hoisted to registers and reused by 3 tiles.
// ---------------------------------------------------------------------------
__global__ void __launch_bounds__(512)
k_xproj(const __bf16* __restrict__ xbf, const __bf16* __restrict__ wkT,
        const float* __restrict__ bias_f, const float* __restrict__ bias_b,
        __bf16* __restrict__ xp) {
  __shared__ __bf16 xs[16][HH];                // 8 KB A tile
  const int dir  = blockIdx.z;
  const int tid  = threadIdx.x;
  const int wave = tid >> 5, lane = tid & 31;
  const int l15  = lane & 15, khalf = lane >> 4;
  const long row0 = (long)blockIdx.x * 16;     // flattened (b,s)
  const int b  = (int)(row0 >> 10);
  const int s0 = (int)(row0 & 1023);

  {                                            // stage A (with bwd reversal)
    const int r  = tid >> 5;                   // 0..15
    const int c8 = (tid & 31) * 8;
    const int sx = dir ? (SS - 1 - (s0 + r)) : (s0 + r);
    *reinterpret_cast<v8bf*>(&xs[r][c8]) =
        *reinterpret_cast<const v8bf*>(xbf + ((size_t)b * SS + sx) * HH + c8);
  }
  __syncthreads();

  const float* b0 = dir ? bias_b : bias_f;     // gru_bias[0]
  const size_t outb = ((size_t)dir * BB + b) * (size_t)SS * H3;

  v16bf afr[8];                                // A fragments, reused by 3 N-tiles
#pragma unroll
  for (int kb = 0; kb < 8; ++kb) afr[kb] = load_a_frag(&xs[l15][0], kb * 32, khalf);

#pragma unroll
  for (int i = 0; i < 3; ++i) {
    const int nt  = wave * 3 + i;              // 0..47
    const int col = nt * 16 + l15;
    const __bf16* bcol = wkT + (size_t)dir * H3 * HH + (size_t)col * HH;
    v8f acc = splat8(b0[col]);
#pragma unroll
    for (int kb = 0; kb < 8; ++kb) {
      acc = wmma_bf16(afr[kb], load_b_frag(bcol, kb * 32, khalf), acc);
    }
#pragma unroll
    for (int g = 0; g < 8; ++g) {
      int m = g + khalf * 8;
      xp[outb + (size_t)(s0 + m) * H3 + col] = (__bf16)acc[g];
    }
  }
}

// ---------------------------------------------------------------------------
// Stage 2: GRU scan (batch-diagonal -> 8 independent workgroups).
// Per step: hp = h @ Wr + b1 via WMMA, then elementwise gates.
// xp[t] slices double-buffered into LDS with global_load_async_to_lds_b128
// (overlaps the WMMA phase; drained with s_wait_asynccnt). First 256 Wr
// columns held in LDS. A fragments hoisted to registers once per step.
// Static LDS = 248 KB (<= 320 KB/WG on CDNA5).
// ---------------------------------------------------------------------------
__global__ void __launch_bounds__(256)
k_gru_scan(const __bf16* __restrict__ xp, const __bf16* __restrict__ wrT,
           const float* __restrict__ bias_f, const float* __restrict__ bias_b,
           __bf16* __restrict__ cbf) {
  __shared__ float  hp[16][H3];                // 48 KB
  __shared__ float  hcur[16][HH];              // 16 KB
  __shared__ __bf16 hbf[16][HH];               //  8 KB
  __shared__ __bf16 xpl[2][16][H3];            // 48 KB (async staged xp[t])
  __shared__ __bf16 wrl[256][HH];              // 128 KB (Wr cols 0..255)
  const int dir = blockIdx.y;
  const int bg  = blockIdx.x;                  // batch rows bg*16 .. +15
  const __bf16* wr = wrT + (size_t)dir * H3 * HH;
  const float* b1 = (dir ? bias_b : bias_f) + H3;   // gru_bias[1]
  const int tid = threadIdx.x, wave = tid >> 5, lane = tid & 31;
  const int l15 = lane & 15, khalf = lane >> 4;

  for (int i = tid; i < 16 * HH; i += 256) {
    hcur[i >> 8][i & 255] = 0.0f;
    hbf[i >> 8][i & 255]  = (__bf16)0.0f;
  }
  // stage Wr columns 0..255 (flat copy; wr is column-major [col][256])
  for (int ch = tid; ch < 256 * HH / 8; ch += 256) {
    reinterpret_cast<v8bf*>(&wrl[0][0])[ch] = reinterpret_cast<const v8bf*>(wr)[ch];
  }
  // pre-splat recurrent bias (loop-invariant across all 1024 steps)
  float b1v[6];
#pragma unroll
  for (int i = 0; i < 6; ++i) b1v[i] = b1[(wave * 6 + i) * 16 + l15];

  // prologue: async-stage xp[0] into buffer 0 (6 x B128 per thread)
  {
#pragma unroll
    for (int j = 0; j < 6; ++j) {
      const int c  = tid + j * 256;            // 0..1535 chunks of 16 B
      const int r  = c / 96;
      const int cb = (c % 96) * 16;
      const char* gp = (const char*)xp +
          ((((size_t)dir * BB + bg * 16 + r) * SS + 0) * H3) * 2 + cb;
      unsigned la = (unsigned)(uintptr_t)(&xpl[0][r][0]) + cb;
      asm volatile("global_load_async_to_lds_b128 %0, %1, off" :: "v"(la), "v"(gp) : "memory");
    }
  }
  __syncthreads();

  for (int t = 0; t < SS; ++t) {
    // issue async stage of xp[t+1] (overlaps WMMA phase below)
    const int tn = t + 1;
    if (tn < SS) {
      const int buf = tn & 1;
#pragma unroll
      for (int j = 0; j < 6; ++j) {
        const int c  = tid + j * 256;
        const int r  = c / 96;
        const int cb = (c % 96) * 16;
        const char* gp = (const char*)xp +
            ((((size_t)dir * BB + bg * 16 + r) * SS + tn) * H3) * 2 + cb;
        unsigned la = (unsigned)(uintptr_t)(&xpl[buf][r][0]) + cb;
        asm volatile("global_load_async_to_lds_b128 %0, %1, off" :: "v"(la), "v"(gp) : "memory");
      }
    }
    // phase 1: hp = hbf @ Wr + b1   (48 N-tiles over 8 waves)
    // A fragments are invariant across the 6 N-tiles: hoist to registers so
    // the inner loop is pure B-load -> WMMA.
    v16bf afr[8];
#pragma unroll
    for (int kb = 0; kb < 8; ++kb) afr[kb] = load_a_frag(&hbf[l15][0], kb * 32, khalf);
#pragma unroll
    for (int i = 0; i < 6; ++i) {
      const int nt = wave * 6 + i;
      const int col = nt * 16 + l15;
      v8f acc = splat8(b1v[i]);
      const __bf16* bcol = (nt < 16) ? (const __bf16*)&wrl[col][0]
                                     : (wr + (size_t)col * HH);
#pragma unroll
      for (int kb = 0; kb < 8; ++kb) {
        acc = wmma_bf16(afr[kb], load_b_frag(bcol, kb * 32, khalf), acc);
      }
#pragma unroll
      for (int g = 0; g < 8; ++g) hp[g + khalf * 8][col] = acc[g];
    }
    // drain this wave's async batch for step t (loads retire in order:
    // <=6 outstanding implies step-t's 6 are complete even with t+1 in flight)
    if (tn < SS) asm volatile("s_wait_asynccnt 0x6" ::: "memory");
    else         asm volatile("s_wait_asynccnt 0x0" ::: "memory");
    __syncthreads();
    // phase 2: gates (reset_after=True), reading staged xp from LDS
    const __bf16* xt = &xpl[t & 1][0][0];
    for (int i = tid; i < 16 * HH; i += 256) {
      const int r = i >> 8, n = i & 255;
      const __bf16* xr_row = xt + (size_t)r * H3;
      const float xz = (float)xr_row[n];
      const float xr = (float)xr_row[HH + n];
      const float xh = (float)xr_row[2 * HH + n];
      const float hz = hp[r][n], hr = hp[r][HH + n], hn = hp[r][2 * HH + n];
      const float z  = sigmoidf_(xz + hz);
      const float rr = sigmoidf_(xr + hr);
      const float hh = tanhf(xh + rr * hn);
      const float hnew = z * hcur[r][n] + (1.0f - z) * hh;
      hcur[r][n] = hnew;
      hbf[r][n]  = (__bf16)hnew;
    }
    __syncthreads();
  }

  for (int i = tid; i < 16 * HH; i += 256) {
    const int r = i >> 8, n = i & 255;
    cbf[(size_t)(bg * 16 + r) * (2 * HH) + dir * HH + n] = (__bf16)hcur[r][n];
  }
}

// ---------------------------------------------------------------------------
// Stage 3a: a = c @ Wa   ([64,512] x [512,65536] -> abf[b][n][e] bf16)
// ---------------------------------------------------------------------------
__global__ void k_hyper_a(const __bf16* __restrict__ cbf, const __bf16* __restrict__ waT,
                          __bf16* __restrict__ abf) {
  const int wave = threadIdx.x >> 5, lane = threadIdx.x & 31;
  const int l15 = lane & 15, khalf = lane >> 4;
  const int mt = blockIdx.x;                       // 0..3
  const int nt = blockIdx.y * 8 + wave;            // 0..4095
  const int col = nt * 16 + l15;
  const __bf16* arow = cbf + (size_t)(mt * 16 + l15) * (2 * HH);
  const __bf16* bcol = waT + (size_t)col * (2 * HH);
  v8f acc = splat8(0.0f);
#pragma unroll
  for (int kb = 0; kb < 16; ++kb) {
    v16bf a = load_a_frag(arow, kb * 32, khalf);
    v16bf w = load_b_frag(bcol, kb * 32, khalf);
    acc = wmma_bf16(a, w, acc);
  }
#pragma unroll
  for (int g = 0; g < 8; ++g) {
    int m = mt * 16 + g + khalf * 8;
    abf[(size_t)m * 65536 + col] = (__bf16)acc[g];
  }
}

// ---------------------------------------------------------------------------
// Stage 3b: per-sample filters, written pre-transposed for the conv GEMM:
//   filtT[b][f_out][k*256+c] = sum_e a[b,f_out,e] * Wb[e, c*K+k]
// ---------------------------------------------------------------------------
__global__ void k_filters(const __bf16* __restrict__ abf, const __bf16* __restrict__ wbT,
                          __bf16* __restrict__ filtT) {
  const int wave = threadIdx.x >> 5, lane = threadIdx.x & 31;
  const int l15 = lane & 15, khalf = lane >> 4;
  const int b  = blockIdx.z;
  const int ntile = blockIdx.x;                    // rows n: 0..15
  const int ft = blockIdx.y * 8 + wave;            // 0..111
  const int f  = ft * 16 + l15;                    // 0..1791 (== m*K + k)
  const __bf16* arow = abf + (size_t)b * 65536 + (size_t)(ntile * 16 + l15) * HH;
  const __bf16* bcol = wbT + (size_t)f * HH;
  v8f acc = splat8(0.0f);
#pragma unroll
  for (int kb = 0; kb < 8; ++kb) {
    v16bf a = load_a_frag(arow, kb * 32, khalf);
    v16bf w = load_b_frag(bcol, kb * 32, khalf);
    acc = wmma_bf16(a, w, acc);
  }
  const int k = f % KK, m = f / KK;
  const size_t base = (size_t)b * NFF * KC;
#pragma unroll
  for (int g = 0; g < 8; ++g) {
    int n = ntile * 16 + g + khalf * 8;
    filtT[base + (size_t)n * KC + k * HH + m] = (__bf16)acc[g];
  }
}

// ---------------------------------------------------------------------------
// Stage 4: per-sample conv as im2col GEMM [1024,1792]x[1792,256] + bias + ReLU.
// grid (64, 1, 64), block 512 (16 waves = 16 f-tiles). Padded x rows (22x256,
// 11 KB) staged once in LDS with zero-filled SAME-padding -> branch-free
// WMMA inner loop and 16x less A traffic.
// ---------------------------------------------------------------------------
__global__ void __launch_bounds__(512)
k_conv(const __bf16* __restrict__ xbf, const __bf16* __restrict__ filtT,
       const float* __restrict__ cbias, float* __restrict__ y) {
  __shared__ __bf16 xs[22][HH];                    // rows st*16-3 .. st*16+18
  const int tid = threadIdx.x;
  const int wave = tid >> 5, lane = tid & 31;
  const int l15 = lane & 15, khalf = lane >> 4;
  const int b  = blockIdx.z;
  const int st = blockIdx.x;                       // s tile 0..63
  const __bf16* xb = xbf + (size_t)b * SS * HH;

  for (int ch = tid; ch < 22 * 32; ch += 512) {    // 16 B chunks
    const int r  = ch >> 5;
    const int c8 = (ch & 31) * 8;
    const int sx = st * 16 + r - 3;
    v8bf v = (sx >= 0 && sx < SS)
                 ? *reinterpret_cast<const v8bf*>(xb + (size_t)sx * HH + c8)
                 : zero8bf();
    *reinterpret_cast<v8bf*>(&xs[r][c8]) = v;
  }
  __syncthreads();

  const int f = wave * 16 + l15;                   // 0..255
  const __bf16* fcol = filtT + (size_t)b * NFF * KC + (size_t)f * KC;
  v8f acc = splat8(cbias[f]);
#pragma unroll 8
  for (int kb = 0; kb < 56; ++kb) {
    const int k  = kb >> 3;                        // tap 0..6
    const int c0 = (kb & 7) * 32;
    v16bf a = load_a_frag(&xs[l15 + k][0], c0, khalf);
    v16bf w = load_b_frag(fcol, kb * 32, khalf);
    acc = wmma_bf16(a, w, acc);
  }
  const size_t ybase = (size_t)b * SS * NFF;
#pragma unroll
  for (int g = 0; g < 8; ++g) {
    int s = st * 16 + g + khalf * 8;
    float v = acc[g];
    y[ybase + (size_t)s * NFF + f] = v > 0.0f ? v : 0.0f;
  }
}

// ---------------------------------------------------------------------------
// Stage 5: MaxPool1D(pool=3, stride=3, VALID) -> out [64,341,256]
// ---------------------------------------------------------------------------
__global__ void k_pool(const float* __restrict__ y, float* __restrict__ out) {
  const long n = (long)BB * POOLED * NFF;
  long i = (long)blockIdx.x * blockDim.x + threadIdx.x;
  if (i >= n) return;
  const int f = (int)(i % NFF);
  const long t = i / NFF;
  const int p = (int)(t % POOLED);
  const int b = (int)(t / POOLED);
  const float* yp = y + ((size_t)b * SS + (size_t)p * 3) * NFF + f;
  out[i] = fmaxf(fmaxf(yp[0], yp[NFF]), yp[2 * NFF]);
}

// ---------------------------------------------------------------------------
// Host side
// ---------------------------------------------------------------------------
extern "C" void kernel_launch(void* const* d_in, const int* in_sizes, int n_in,
                              void* d_out, int out_size, void* d_ws, size_t ws_size,
                              hipStream_t stream) {
  const float* x      = (const float*)d_in[0];
  const float* wk_f   = (const float*)d_in[2];
  const float* wr_f   = (const float*)d_in[3];
  const float* bias_f = (const float*)d_in[4];
  const float* wk_b   = (const float*)d_in[5];
  const float* wr_b   = (const float*)d_in[6];
  const float* bias_b = (const float*)d_in[7];
  const float* Wa     = (const float*)d_in[8];
  const float* Wb     = (const float*)d_in[9];
  const float* cbias  = (const float*)d_in[10];
  float* out = (float*)d_out;
  (void)in_sizes; (void)n_in; (void)out_size; (void)ws_size;

  // workspace layout (bytes, 256-aligned)
  char* ws = (char*)d_ws;
  size_t o = 0;
  auto take = [&](size_t bytes) { char* p = ws + o; o = (o + bytes + 255) & ~(size_t)255; return p; };
  __bf16* xbf   = (__bf16*)take((size_t)BB * SS * HH * 2);            // 33.6 MB
  __bf16* xp    = (__bf16*)take((size_t)2 * BB * SS * H3 * 2);        // 201 MB
  __bf16* wkT   = (__bf16*)take((size_t)2 * H3 * HH * 2);
  __bf16* wrT   = (__bf16*)take((size_t)2 * H3 * HH * 2);
  __bf16* waT   = (__bf16*)take((size_t)65536 * 512 * 2);             // 67 MB
  __bf16* wbT   = (__bf16*)take((size_t)KC * HH * 2);
  __bf16* cbf   = (__bf16*)take((size_t)BB * 2 * HH * 2);
  __bf16* abf   = (__bf16*)take((size_t)BB * 65536 * 2);              // 8.4 MB
  __bf16* filtT = (__bf16*)take((size_t)BB * NFF * KC * 2);           // 58.7 MB
  float*  ybuf  = (float*) take((size_t)BB * SS * NFF * 4);           // 67 MB

  // prep
  {
    long n = (long)BB * SS * HH;
    k_cast_bf16<<<dim3((unsigned)((n + 255) / 256)), dim3(256), 0, stream>>>(x, xbf, n);
  }
  dim3 tb(16, 16);
  k_transpose_bf16<<<dim3(H3 / 16, HH / 16), tb, 0, stream>>>(wk_f, wkT, HH, H3);
  k_transpose_bf16<<<dim3(H3 / 16, HH / 16), tb, 0, stream>>>(wk_b, wkT + (size_t)H3 * HH, HH, H3);
  k_transpose_bf16<<<dim3(H3 / 16, HH / 16), tb, 0, stream>>>(wr_f, wrT, HH, H3);
  k_transpose_bf16<<<dim3(H3 / 16, HH / 16), tb, 0, stream>>>(wr_b, wrT + (size_t)H3 * HH, HH, H3);
  k_transpose_bf16<<<dim3(65536 / 16, 512 / 16), tb, 0, stream>>>(Wa, waT, 512, 65536);
  k_transpose_bf16<<<dim3(KC / 16, HH / 16), tb, 0, stream>>>(Wb, wbT, HH, KC);

  // stage 1: input projections (both directions)
  k_xproj<<<dim3(BB * SS / 16, 1, 2), dim3(512), 0, stream>>>(xbf, wkT, bias_f, bias_b, xp);

  // stage 2: batch-partitioned GRU scans (8 independent workgroups)
  k_gru_scan<<<dim3(BB / 16, 2), dim3(256), 0, stream>>>(xp, wrT, bias_f, bias_b, cbf);

  // stage 3: hypernetwork
  k_hyper_a<<<dim3(4, 512), dim3(256), 0, stream>>>(cbf, waT, abf);
  k_filters<<<dim3(16, 14, BB), dim3(256), 0, stream>>>(abf, wbT, filtT);

  // stage 4: per-sample conv (+bias, ReLU)
  k_conv<<<dim3(SS / 16, 1, BB), dim3(512), 0, stream>>>(xbf, filtT, cbias, ybuf);

  // stage 5: maxpool
  {
    long n = (long)BB * POOLED * NFF;
    k_pool<<<dim3((unsigned)((n + 255) / 256)), dim3(256), 0, stream>>>(ybuf, out);
  }
}
